// SelfAttention_50586124812319
// MI455X (gfx1250) — matrix operand order repaired and checked by reading, not compiled, not run on hardware
//
#include <hip/hip_runtime.h>
#include <hip/hip_bf16.h>

// Problem dims (fixed by the reference)
#define BB   8
#define CC   256
#define CQK  32
#define NN   4096   // H*W = 64*64

typedef _Float16 h8  __attribute__((ext_vector_type(8)));
typedef _Float16 h16 __attribute__((ext_vector_type(16)));
typedef float    f8  __attribute__((ext_vector_type(8)));
typedef unsigned int u32x4 __attribute__((ext_vector_type(4)));
typedef int          i32x4 __attribute__((ext_vector_type(4)));
typedef int          i32x8 __attribute__((ext_vector_type(8)));

#define HAVE_TDM __has_builtin(__builtin_amdgcn_tensor_load_to_lds)

// Padded LDS row stride for the V tile: 64B row + 16B TDM pad = 80B (20 banks,
// conflict-free for 16-lane row-strided b128 reads, still 16B aligned).
#define VROW 40   // in f16 units

// ---------------------------------------------------------------------------
// Kernel 1: fused projections  f = wq x, g = wk x, hh = wv x  (f32 -> f16)
// ---------------------------------------------------------------------------
__global__ __launch_bounds__(256) void proj_kernel(
    const float* __restrict__ x,
    const float* __restrict__ wq,
    const float* __restrict__ wk,
    const float* __restrict__ wv,
    _Float16* __restrict__ fT,
    _Float16* __restrict__ gT,
    _Float16* __restrict__ hh)
{
    __shared__ float wrow[CC];
    const int blk = blockIdx.x;
    const int b   = blk / 320;
    const int o   = blk % 320;
    const int t   = threadIdx.x;

    const float* w = (o < 32) ? (wq + (size_t)o * CC)
                   : (o < 64) ? (wk + (size_t)(o - 32) * CC)
                              : (wv + (size_t)(o - 64) * CC);
    wrow[t] = w[t];
    __syncthreads();

    const float* xb = x + (size_t)b * CC * NN;

    float acc[16];
#pragma unroll
    for (int j = 0; j < 16; ++j) acc[j] = 0.0f;

    for (int c = 0; c < CC; ++c) {
        const float wc = wrow[c];
        const float* xr = xb + (size_t)c * NN + t;
#pragma unroll
        for (int j = 0; j < 16; ++j)
            acc[j] = fmaf(wc, xr[j * 256], acc[j]);
    }

    if (o < 32) {
        _Float16* dst = fT + (size_t)b * NN * CQK + o;
#pragma unroll
        for (int j = 0; j < 16; ++j)
            dst[(size_t)(t + j * 256) * CQK] = (_Float16)acc[j];
    } else if (o < 64) {
        _Float16* dst = gT + (size_t)b * NN * CQK + (o - 32);
#pragma unroll
        for (int j = 0; j < 16; ++j)
            dst[(size_t)(t + j * 256) * CQK] = (_Float16)acc[j];
    } else {
        _Float16* dst = hh + (size_t)b * CC * NN + (size_t)(o - 64) * NN;
#pragma unroll
        for (int j = 0; j < 16; ++j)
            dst[t + j * 256] = (_Float16)acc[j];
    }
}

#if HAVE_TDM
// TDM: DMA one V key-block (256 rows x 64B) from global into LDS, padding
// 16B after each 64B row (LDS row stride 80B).  D# per ISA 08_async_tensor §8.
// This toolchain's builtin takes 6 args (g0, g1, g2, g3, extra group, cpol).
__device__ __forceinline__ void tdm_load_vtile(const _Float16* gsrc, unsigned lds_off)
{
    const unsigned long long ga = (unsigned long long)(uintptr_t)gsrc;
    u32x4 g0;
    g0[0] = 1u;                                            // count=1 (valid user D#)
    g0[1] = lds_off;                                       // lds_addr (bytes)
    g0[2] = (unsigned)(ga & 0xFFFFFFFFull);                // global_addr[31:0]
    g0[3] = (unsigned)((ga >> 32) & 0x01FFFFFFull)         // global_addr[56:32]
          | 0x80000000u;                                   // type = 2 ("image")
    i32x8 g1;
    g1[0] = (int)((3u << 16)        // data_size = 8 bytes
                | (1u << 20)        // pad_enable
                | (3u << 22)        // pad_interval: 16 DWORDs (one 64B row)
                | (3u << 25));      // pad_amount:   4 DWORDs (16B)
    g1[1] = (int)(1024u << 16);     // tensor_dim0 = 1024 units (N*2/8 bytes)
    g1[2] = (int)(256u  << 16);     // tensor_dim1 = 256 rows
    g1[3] = (int)(8u    << 16);     // tile_dim0   = 8 units (64B of keys)
    g1[4] = (int)256;               // tile_dim1   = 256 rows, tile_dim2 = 0
    g1[5] = (int)1024;              // tensor_dim0_stride = 1024 units (8KB row)
    g1[6] = 0;
    g1[7] = 0;
    const i32x4 z4 = {0, 0, 0, 0};              // 2-D tile: groups 2/3 unused
    const i32x8 z8 = {0, 0, 0, 0, 0, 0, 0, 0};  // unused trailing group
    __builtin_amdgcn_tensor_load_to_lds(g0, g1, z4, z4, z8, 0);
}
#endif

// ---------------------------------------------------------------------------
// Kernel 2: fused attention.  One wave32 per 16-query tile, 8 waves/block
// sharing one batch; the 256x32 V key-block is TDM-DMA'd into LDS once per
// block (double buffered), K fragments stream from global, P is re-laid out
// D-fragment -> B-fragment through a 1KB/wave LDS tile.
// ---------------------------------------------------------------------------
__global__ __launch_bounds__(256) void attn_kernel(
    const float* __restrict__ x,
    const float* __restrict__ gamma,
    const _Float16* __restrict__ fT,
    const _Float16* __restrict__ gT,
    const _Float16* __restrict__ hh,
    float* __restrict__ out)
{
#if HAVE_TDM
    __shared__ _Float16 Vbuf[2][CC * VROW];   // 2 x 20KB padded V tiles
#endif
    __shared__ _Float16 Pl[8][16][32];        // per-wave P staging (8KB)

    const int lane = threadIdx.x & 31;
    const int wid  = threadIdx.x >> 5;
    const int b    = blockIdx.x >> 5;                       // 32 blocks per batch
    const int n0   = (((blockIdx.x & 31) << 3) + wid) * 16; // this wave's queries

    const int lo16 = lane & 15;              // col (B/C/D) or row (A) index
    const int hi   = lane >> 4;              // lane-half select

    const _Float16* fbase = fT + (size_t)b * NN * CQK;
    const _Float16* gbase = gT + (size_t)b * NN * CQK;
    const _Float16* vbase = hh + (size_t)b * CC * NN;

    // Q B-fragment (32x16): lane holds K = hi*16 .. hi*16+15 at col n0+lo16
    const h16 qf = *(const h16*)(gbase + (size_t)(n0 + lo16) * CQK + hi * 16);

    f8 accO[16];
#pragma unroll
    for (int ct = 0; ct < 16; ++ct)
#pragma unroll
        for (int r = 0; r < 8; ++r) accO[ct][r] = 0.0f;

    float cs = 0.0f;   // per-lane partial exp-sum for column n0+lo16
    f8 zc;
#pragma unroll
    for (int r = 0; r < 8; ++r) zc[r] = 0.0f;

    const int NMB = NN / 32;   // 128 key blocks

#if HAVE_TDM
    // Prologue: DMA key-block 0 into buffer 0 (one issue per block; TDM
    // ignores EXEC, so gate on wave id and publish via barrier).
    if (wid == 0) {
        tdm_load_vtile(vbase, (unsigned)(uintptr_t)&Vbuf[0][0]);
        __builtin_amdgcn_s_wait_tensorcnt(0);
    }
    __syncthreads();
#endif

    for (int mb = 0; mb < NMB; ++mb) {
        const int m0  = mb * 32;
#if HAVE_TDM
        const int cur = mb & 1;
        if (wid == 0 && mb + 1 < NMB)   // async fill of the other buffer
            tdm_load_vtile(vbase + (size_t)(m0 + 32),
                           (unsigned)(uintptr_t)&Vbuf[cur ^ 1][0]);
#endif

        // ---- A fragments of K^T: two 16x32 tiles (keys m0..m0+15, +16..31)
        const _Float16* fr0 = fbase + (size_t)(m0 + lo16) * CQK;
        const _Float16* fr1 = fbase + (size_t)(m0 + 16 + lo16) * CQK;
        h8 a0lo = *(const h8*)(fr0 + hi * 16);
        h8 a0hi = *(const h8*)(fr0 + hi * 16 + 8);
        h8 a1lo = *(const h8*)(fr1 + hi * 16);
        h8 a1hi = *(const h8*)(fr1 + hi * 16 + 8);
        h16 a0, a1;
#pragma unroll
        for (int i = 0; i < 8; ++i) {
            a0[i] = a0lo[i]; a0[8 + i] = a0hi[i];
            a1[i] = a1lo[i]; a1[8 + i] = a1hi[i];
        }

        if (m0 + 32 < NN)   // speculative prefetch of next key rows
            __builtin_prefetch(fbase + (size_t)(m0 + 32 + lo16) * CQK, 0, 1);

        // ---- S tiles (16 keys x 16 queries each), f32 accum
        f8 s0 = __builtin_amdgcn_wmma_f32_16x16x32_f16(false, a0, false, qf,
                                                       (short)0, zc, false, false);
        f8 s1 = __builtin_amdgcn_wmma_f32_16x16x32_f16(false, a1, false, qf,
                                                       (short)0, zc, false, false);

        // ---- P = exp(S); accumulate per-column partial sums
        h8 p0, p1;
#pragma unroll
        for (int r = 0; r < 8; ++r) {
            float e0 = __expf(s0[r]);
            float e1 = __expf(s1[r]);
            cs += e0 + e1;
            p0[r] = (_Float16)e0;
            p1[r] = (_Float16)e1;
        }

        // ---- Re-layout P (D-fragment -> B-fragment) through LDS
        *(h8*)&Pl[wid][lo16][hi * 8]      = p0;   // S0 rows  0..15
        *(h8*)&Pl[wid][lo16][16 + hi * 8] = p1;   // S1 rows 16..31
        asm volatile("s_wait_dscnt 0" ::: "memory");   // intra-wave DS ordering
        const h16 pf = *(const h16*)&Pl[wid][lo16][hi * 16];

        // ---- O += V_tile (16c x 32m) * P (32m x 16n), 16 channel tiles
#pragma unroll
        for (int ct = 0; ct < 16; ++ct) {
#if HAVE_TDM
            const _Float16* vr = &Vbuf[cur][(ct * 16 + lo16) * VROW];
#else
            const _Float16* vr = vbase + (size_t)(ct * 16 + lo16) * NN + m0;
#endif
            h8 vlo = *(const h8*)(vr + hi * 16);
            h8 vhi = *(const h8*)(vr + hi * 16 + 8);
            h16 av;
#pragma unroll
            for (int i = 0; i < 8; ++i) { av[i] = vlo[i]; av[8 + i] = vhi[i]; }
            accO[ct] = __builtin_amdgcn_wmma_f32_16x16x32_f16(
                false, av, false, pf, (short)0, accO[ct], false, false);
        }

#if HAVE_TDM
        if (wid == 0 && mb + 1 < NMB)
            __builtin_amdgcn_s_wait_tensorcnt(0);  // next buffer filled
        __syncthreads();                           // publish / protect buffers
#endif
    }

    // ---- Epilogue: combine the two lane-halves' column sums, normalize,
    //      scale by gamma, add residual, store f32.
    const float total = cs + __shfl_xor(cs, 16, 32);
    const float scale = gamma[0] / total;

    const float* xb = x   + (size_t)b * CC * NN;
    float*       ob = out + (size_t)b * CC * NN;
#pragma unroll
    for (int ct = 0; ct < 16; ++ct) {
#pragma unroll
        for (int r = 0; r < 8; ++r) {
            const int c = ct * 16 + hi * 8 + r;       // D layout: M = hi*8 + r
            const size_t idx = (size_t)c * NN + (n0 + lo16);
            ob[idx] = fmaf(scale, accO[ct][r], xb[idx]);
        }
    }
}

// ---------------------------------------------------------------------------
extern "C" void kernel_launch(void* const* d_in, const int* in_sizes, int n_in,
                              void* d_out, int out_size, void* d_ws, size_t ws_size,
                              hipStream_t stream) {
    const float* x     = (const float*)d_in[0];
    const float* wq    = (const float*)d_in[1];
    const float* wk    = (const float*)d_in[2];
    const float* wv    = (const float*)d_in[3];
    const float* gamma = (const float*)d_in[4];
    float* out = (float*)d_out;

    _Float16* fT = (_Float16*)d_ws;                       // [B][N][32]
    _Float16* gT = fT + (size_t)BB * NN * CQK;            // [B][N][32]
    _Float16* hh = gT + (size_t)BB * NN * CQK;            // [B][C][N]

    // 320 output rows (32 f + 32 g + 256 hh) per batch
    proj_kernel<<<BB * 320, 256, 0, stream>>>(x, wq, wk, wv, fT, gT, hh);

    // 8 waves per block share one batch + one TDM-staged V key-block
    attn_kernel<<<(BB * (NN / 16)) / 8, 256, 0, stream>>>(x, gamma, fT, gT, hh, out);
}
